// DecoderBlock_15805479649978
// MI455X (gfx1250) — compile-verified, matrix operand order checked by reading, not run on hardware
//
#include <hip/hip_runtime.h>

#define B_   8
#define L_   1024
#define M_   1024
#define H_   16
#define D_   1024
#define DK_  64
#define DFF_ 4096
#define EPS_ 1e-5f
#define NT_  (B_ * L_)   // 8192 token rows

typedef __attribute__((ext_vector_type(16))) __bf16 v16bf;
typedef __attribute__((ext_vector_type(8)))  float  v8f;
typedef __attribute__((ext_vector_type(4)))  unsigned int u32x4;
typedef __attribute__((ext_vector_type(8)))  int i32x8;
typedef __attribute__((ext_vector_type(4)))  int i32x4;

#if __has_builtin(__builtin_amdgcn_tensor_load_to_lds)
#define HAVE_TDM 1
#else
#define HAVE_TDM 0
#endif

union Frag16 {
  v16bf v;
  uint4 u[2];
};

__device__ __forceinline__ unsigned short f32_to_bf16(float f) {
  unsigned int u = __float_as_uint(f);
  u += 0x7FFFu + ((u >> 16) & 1u);           // round-to-nearest-even
  return (unsigned short)(u >> 16);
}

// 16x32 bf16 fragment (A layout; identical B layout when the source matrix is
// stored transposed, i.e. X[n][k] row-major). ISA 7.12.2:
//   lanes 0-15  -> row = lane,    K {0..7, 16..23}
//   lanes 16-31 -> row = lane-16, K {8..15, 24..31}
__device__ __forceinline__ v16bf load_frag(const unsigned short* p, int ld) {
  const int lane = threadIdx.x & 31;
  const int m    = lane & 15;
  const int kb   = (lane & 16) ? 8 : 0;
  const unsigned short* base = p + (size_t)m * (size_t)ld + kb;
  Frag16 f;
  f.u[0] = *reinterpret_cast<const uint4*>(base);        // K kb..kb+7
  f.u[1] = *reinterpret_cast<const uint4*>(base + 16);   // K kb+16..kb+23
  return f.v;
}

__device__ __forceinline__ v8f wmma_bf16(v16bf a, v16bf b, v8f c) {
  return __builtin_amdgcn_wmma_f32_16x16x32_bf16(false, a, false, b,
                                                 (short)0, c, false, false);
}

__device__ __forceinline__ void wait_ds0() {
#if __has_builtin(__builtin_amdgcn_s_wait_dscnt)
  __builtin_amdgcn_s_wait_dscnt(0);
#else
  asm volatile("s_wait_dscnt 0" ::: "memory");
#endif
}

__device__ __forceinline__ void wait_tensor0() {
#if __has_builtin(__builtin_amdgcn_s_wait_tensorcnt)
  __builtin_amdgcn_s_wait_tensorcnt(0);
#else
  asm volatile("s_wait_tensorcnt 0" ::: "memory");
#endif
}

#if HAVE_TDM
// One-wave TDM: DMA a (tile_h x tile_w) bf16 tile (row stride = stride_elems)
// from global into LDS at lds_addr, contiguous row-major. D# per ISA ch.8:
// group0 {count=1, lds_addr, global_addr, type=2}; group1 {data_size=2B,
// tensor==tile dims, stride0}. Groups 2/3 zero (2-D tile).
__device__ __forceinline__ void tdm_load_2d(unsigned lds_addr, const void* gaddr,
                                            unsigned tile_w, unsigned tile_h,
                                            unsigned long long stride_elems) {
  unsigned long long ga = (unsigned long long)(size_t)gaddr;
  u32x4 g0;
  g0[0] = 1u;                                           // count=1 (valid D#)
  g0[1] = lds_addr;                                     // LDS byte address
  g0[2] = (unsigned)(ga & 0xFFFFFFFFu);                 // global_addr[31:0]
  g0[3] = (unsigned)((ga >> 32) & 0x01FFFFFFu) | (2u << 30);  // [56:32] | type=2
  i32x8 g1;
  g1[0] = (int)(1u << 16);                              // data_size=1 (2 bytes)
  g1[1] = (int)((tile_w & 0xFFFFu) << 16);              // tensor_dim0[15:0]
  g1[2] = (int)(((tile_w >> 16) & 0xFFFFu) | ((tile_h & 0xFFFFu) << 16));
  g1[3] = (int)(((tile_h >> 16) & 0xFFFFu) | (tile_w << 16));   // tile_dim0
  g1[4] = (int)(tile_h & 0xFFFFu);                      // tile_dim1 (tile_dim2=0)
  g1[5] = (int)(unsigned)(stride_elems & 0xFFFFFFFFull);        // stride0[31:0]
  g1[6] = (int)(unsigned)((stride_elems >> 32) & 0xFFFFull);    // stride0[47:32]
  g1[7] = 0;
  i32x4 z4 = {0, 0, 0, 0};
  i32x8 z8 = {0, 0, 0, 0, 0, 0, 0, 0};
  __builtin_amdgcn_tensor_load_to_lds(g0, g1, z4, z4, z8, 0);
}
#endif

// ---------------------------------------------------------------------------
__global__ __launch_bounds__(256) void cvt_kernel(const float* __restrict__ in,
                                                  unsigned short* __restrict__ out,
                                                  size_t n) {
  size_t i = (size_t)blockIdx.x * 256 + threadIdx.x;
  if (i < n) out[i] = f32_to_bf16(in[i]);
}

// fp32 [R][C] -> bf16 transposed [C][R]
__global__ __launch_bounds__(256) void cvt_t_kernel(const float* __restrict__ in,
                                                    unsigned short* __restrict__ out,
                                                    int R, int C) {
  size_t idx = (size_t)blockIdx.x * 256 + threadIdx.x;
  if (idx >= (size_t)R * C) return;
  int r = (int)(idx / C);
  int c = (int)(idx % C);
  out[(size_t)c * R + r] = f32_to_bf16(in[idx]);
}

// [H, D, DK] fp32 -> W^T bf16 [H*DK][D]  (so GEMM B operand is [N][K] row-major)
__global__ __launch_bounds__(256) void pack_qkv_kernel(const float* __restrict__ W,
                                                       unsigned short* __restrict__ out) {
  size_t idx = (size_t)blockIdx.x * 256 + threadIdx.x;  // H*D*DK = 1M, exact
  int k = (int)(idx % DK_);
  size_t r = idx / DK_;
  int d = (int)(r % D_);
  int h = (int)(r / D_);
  out[(size_t)(h * DK_ + k) * D_ + d] = f32_to_bf16(W[idx]);
}

// ---------------------------------------------------------------------------
// WMMA bf16 GEMM: C[M,N] = A[M,K] @ W^T[N,K]^T (+bias, opt ReLU).
// Block: 256 thr = 8 waves (4x2), tile 128x128, wave tile 32x64, K-step 32,
// 8 WMMAs per wave per K-step. A/W^T tiles DMA'd to LDS by the TDM when
// available; otherwise fragments stream straight from global (L2 resident).
// SBT: store bf16 output transposed ([N][M]) for the attention V operand.
template <int RELU, int SF32, int SB16, int SBT>
__global__ __launch_bounds__(256)
void gemm_kernel(const unsigned short* __restrict__ A,
                 const unsigned short* __restrict__ WT,
                 const float* __restrict__ bias,
                 float* __restrict__ Cf, unsigned short* __restrict__ Cb,
                 int M, int N, int K) {
  const int tid  = threadIdx.x;
  const int lane = tid & 31;
  const int wid  = tid >> 5;
  const int wm   = wid & 3;
  const int wn   = wid >> 2;
  const int r0   = blockIdx.y * 128 + wm * 32;
  const int c0   = blockIdx.x * 128 + wn * 64;

  v8f acc[2][4] = {};

#if HAVE_TDM
  __shared__ unsigned short AT[128][32];
  __shared__ unsigned short BT[128][32];
  const unsigned at_lds = (unsigned)(size_t)&AT[0][0];
  const unsigned bt_lds = (unsigned)(size_t)&BT[0][0];
#endif

  for (int k0 = 0; k0 < K; k0 += 32) {
#if HAVE_TDM
    __syncthreads();   // previous tile fully consumed
    if (wid == 0)
      tdm_load_2d(at_lds, A + (size_t)(blockIdx.y * 128) * K + k0, 32, 128,
                  (unsigned long long)K);
    else if (wid == 1)
      tdm_load_2d(bt_lds, WT + (size_t)(blockIdx.x * 128) * K + k0, 32, 128,
                  (unsigned long long)K);
    if (wid < 2) wait_tensor0();
    __syncthreads();   // tiles visible to all waves
    v16bf a0 = load_frag(&AT[wm * 32][0], 32);
    v16bf a1 = load_frag(&AT[wm * 32 + 16][0], 32);
    v16bf b0 = load_frag(&BT[wn * 64][0], 32);
    v16bf b1 = load_frag(&BT[wn * 64 + 16][0], 32);
    v16bf b2 = load_frag(&BT[wn * 64 + 32][0], 32);
    v16bf b3 = load_frag(&BT[wn * 64 + 48][0], 32);
#else
    __builtin_prefetch(A + (size_t)r0 * K + k0 + 32, 0, 0);
    v16bf a0 = load_frag(A + (size_t)r0 * K + k0, K);
    v16bf a1 = load_frag(A + (size_t)(r0 + 16) * K + k0, K);
    v16bf b0 = load_frag(WT + (size_t)(c0)      * K + k0, K);
    v16bf b1 = load_frag(WT + (size_t)(c0 + 16) * K + k0, K);
    v16bf b2 = load_frag(WT + (size_t)(c0 + 32) * K + k0, K);
    v16bf b3 = load_frag(WT + (size_t)(c0 + 48) * K + k0, K);
#endif
    acc[0][0] = wmma_bf16(a0, b0, acc[0][0]);
    acc[0][1] = wmma_bf16(a0, b1, acc[0][1]);
    acc[0][2] = wmma_bf16(a0, b2, acc[0][2]);
    acc[0][3] = wmma_bf16(a0, b3, acc[0][3]);
    acc[1][0] = wmma_bf16(a1, b0, acc[1][0]);
    acc[1][1] = wmma_bf16(a1, b1, acc[1][1]);
    acc[1][2] = wmma_bf16(a1, b2, acc[1][2]);
    acc[1][3] = wmma_bf16(a1, b3, acc[1][3]);
  }

  const int rb = (lane & 16) ? 8 : 0;
  const int cl = lane & 15;
#pragma unroll
  for (int tm = 0; tm < 2; ++tm)
#pragma unroll
    for (int tn = 0; tn < 4; ++tn) {
      int gcol = c0 + tn * 16 + cl;
      float bi = bias ? bias[gcol] : 0.f;
#pragma unroll
      for (int i = 0; i < 8; ++i) {
        int grow = r0 + tm * 16 + rb + i;
        float v = acc[tm][tn][i] + bi;
        if (RELU) v = fmaxf(v, 0.f);
        if (SF32) Cf[(size_t)grow * N + gcol] = v;
        if (SB16) Cb[(size_t)grow * N + gcol] = f32_to_bf16(v);
        if (SBT)  Cb[(size_t)gcol * M + grow] = f32_to_bf16(v);  // [N][M]
      }
    }
}

// ---------------------------------------------------------------------------
// Attention per (b,h): S = Q K^T / 8, softmax over ALL keys, mask applied
// AFTER softmax (reference semantics), out = P V.
// Q/K/Z: [B*L, H*DK] bf16; VT: [H*DK][B*Lk] bf16 (V stored transposed by the
// projection GEMM, so PV B-fragments load straight from global, no LDS
// transpose and no block barriers). Block: 128 thr = 4 independent waves,
// one 16-row q tile each; 32-key chunks, 8 WMMAs per chunk.
template <bool MASKED>
__global__ __launch_bounds__(128)
void attn_kernel(const unsigned short* __restrict__ Q,
                 const unsigned short* __restrict__ K,
                 const unsigned short* __restrict__ VT,
                 unsigned short* __restrict__ Z, int Lk) {
  __shared__ unsigned short shP[4][16][32];    // per-wave P tile [q][key]

  const int ld   = H_ * DK_;
  const int b    = blockIdx.z;
  const int h    = blockIdx.y;
  const int wave = threadIdx.x >> 5;
  const int lane = threadIdx.x & 31;
  const int q0   = blockIdx.x * 64 + wave * 16;
  const int rb   = (lane & 16) ? 8 : 0;
  const int cl   = lane & 15;

  const unsigned short* Qb  = Q + (size_t)b * L_ * ld + h * DK_;
  const unsigned short* Kb  = K + (size_t)b * Lk * ld + h * DK_;
  const unsigned short* VTb = VT + (size_t)h * DK_ * NT_ + (size_t)b * Lk;

  v16bf aq0 = load_frag(Qb + (size_t)q0 * ld + 0, ld);   // Q[16 x dk0..31]
  v16bf aq1 = load_frag(Qb + (size_t)q0 * ld + 32, ld);  // Q[16 x dk32..63]

  float mrow[8], drow[8];
  v8f acc[4] = {};
#pragma unroll
  for (int i = 0; i < 8; ++i) { mrow[i] = -1e30f; drow[i] = 0.f; }

  for (int j0 = 0; j0 < Lk; j0 += 32) {
    // scores: two 16x16 C tiles (keys j0..+15, j0+16..+31), contraction dk=64
    v8f s0 = {}, s1 = {};
    v16bf bk;
    bk = load_frag(Kb + (size_t)(j0)      * ld + 0,  ld); s0 = wmma_bf16(aq0, bk, s0);
    bk = load_frag(Kb + (size_t)(j0)      * ld + 32, ld); s0 = wmma_bf16(aq1, bk, s0);
    bk = load_frag(Kb + (size_t)(j0 + 16) * ld + 0,  ld); s1 = wmma_bf16(aq0, bk, s1);
    bk = load_frag(Kb + (size_t)(j0 + 16) * ld + 32, ld); s1 = wmma_bf16(aq1, bk, s1);

    const bool skipPV = MASKED && (j0 > q0 + 15);  // wave-uniform

#pragma unroll
    for (int i = 0; i < 8; ++i) {
      float v0 = s0[i] * 0.125f;     // 1/sqrt(DK)
      float v1 = s1[i] * 0.125f;
      float t = fmaxf(v0, v1);
#pragma unroll
      for (int mk = 1; mk < 16; mk <<= 1) t = fmaxf(t, __shfl_xor(t, mk, 32));
      float mnew  = fmaxf(mrow[i], t);
      float scale = __expf(mrow[i] - mnew);
      float p0 = __expf(v0 - mnew);
      float p1 = __expf(v1 - mnew);
      float rs = p0 + p1;            // denominator counts ALL keys (pre-mask softmax)
#pragma unroll
      for (int mk = 1; mk < 16; mk <<= 1) rs += __shfl_xor(rs, mk, 32);
      drow[i] = drow[i] * scale + rs;
      mrow[i] = mnew;
#pragma unroll
      for (int t4 = 0; t4 < 4; ++t4) acc[t4][i] *= scale;
      if (!skipPV) {
        if (MASKED) {                // mask numerator only
          int qrow = q0 + rb + i;
          if (j0 + cl > qrow)      p0 = 0.f;
          if (j0 + 16 + cl > qrow) p1 = 0.f;
        }
        shP[wave][rb + i][cl]      = f32_to_bf16(p0);
        shP[wave][rb + i][cl + 16] = f32_to_bf16(p1);
      }
    }

    if (!skipPV) {
      wait_ds0();  // wave-local P writes visible before fragment reads
      v16bf ap = load_frag(&shP[wave][0][0], 32);          // P [16 x 32 keys]
#pragma unroll
      for (int t4 = 0; t4 < 4; ++t4) {
        v16bf bv = load_frag(VTb + (size_t)(t4 * 16) * NT_ + j0, NT_);
        acc[t4] = wmma_bf16(ap, bv, acc[t4]);
      }
    }
  }

#pragma unroll
  for (int t4 = 0; t4 < 4; ++t4)
#pragma unroll
    for (int i = 0; i < 8; ++i) {
      float o = acc[t4][i] / drow[i];
      int qrow = q0 + rb + i;
      Z[((size_t)b * L_ + qrow) * ld + h * DK_ + t4 * 16 + cl] = f32_to_bf16(o);
    }
}

// ---------------------------------------------------------------------------
// LayerNorm(y + resid) * g + be ; writes f32 and optional bf16. One row/block.
__global__ __launch_bounds__(256)
void ln_kernel(const float* __restrict__ y, const float* __restrict__ resid,
               const float* __restrict__ g, const float* __restrict__ be,
               float* __restrict__ outf, unsigned short* __restrict__ outb) {
  __shared__ float ssum[8], ssq[8];
  size_t row = blockIdx.x;
  const float* yp = y + row * D_;
  const float* rp = resid + row * D_;
  int t = threadIdx.x;
  float v[4];
  float s = 0.f, sq = 0.f;
#pragma unroll
  for (int i = 0; i < 4; ++i) {
    float x = yp[t * 4 + i] + rp[t * 4 + i];
    v[i] = x; s += x; sq += x * x;
  }
#pragma unroll
  for (int mk = 1; mk < 32; mk <<= 1) {
    s  += __shfl_xor(s, mk, 32);
    sq += __shfl_xor(sq, mk, 32);
  }
  if ((t & 31) == 0) { ssum[t >> 5] = s; ssq[t >> 5] = sq; }
  __syncthreads();
  float ts = 0.f, tq = 0.f;
#pragma unroll
  for (int w = 0; w < 8; ++w) { ts += ssum[w]; tq += ssq[w]; }
  float mean = ts * (1.f / D_);
  float var  = tq * (1.f / D_) - mean * mean;
  float inv  = rsqrtf(var + EPS_);
#pragma unroll
  for (int i = 0; i < 4; ++i) {
    int c = t * 4 + i;
    float o = (v[i] - mean) * inv * g[c] + be[c];
    outf[row * D_ + c] = o;
    if (outb) outb[row * D_ + c] = f32_to_bf16(o);
  }
}

// ---------------------------------------------------------------------------
extern "C" void kernel_launch(void* const* d_in, const int* in_sizes, int n_in,
                              void* d_out, int out_size, void* d_ws, size_t ws_size,
                              hipStream_t stream) {
  const float* x   = (const float*)d_in[0];
  const float* enc = (const float*)d_in[1];
  const float* Wq1 = (const float*)d_in[2];  const float* bq1 = (const float*)d_in[3];
  const float* Wk1 = (const float*)d_in[4];  const float* bk1 = (const float*)d_in[5];
  const float* Wv1 = (const float*)d_in[6];  const float* bv1 = (const float*)d_in[7];
  const float* Wo1 = (const float*)d_in[8];  const float* bo1 = (const float*)d_in[9];
  const float* Wq2 = (const float*)d_in[10]; const float* bq2 = (const float*)d_in[11];
  const float* Wk2 = (const float*)d_in[12]; const float* bk2 = (const float*)d_in[13];
  const float* Wv2 = (const float*)d_in[14]; const float* bv2 = (const float*)d_in[15];
  const float* Wo2 = (const float*)d_in[16]; const float* bo2 = (const float*)d_in[17];
  const float* W1  = (const float*)d_in[18]; const float* b1  = (const float*)d_in[19];
  const float* W2  = (const float*)d_in[20]; const float* b2  = (const float*)d_in[21];
  const float* g1  = (const float*)d_in[22]; const float* be1 = (const float*)d_in[23];
  const float* g2  = (const float*)d_in[24]; const float* be2 = (const float*)d_in[25];
  const float* g3  = (const float*)d_in[26]; const float* be3 = (const float*)d_in[27];
  (void)in_sizes; (void)n_in; (void)out_size; (void)ws_size;

  char* ws = (char*)d_ws;
  size_t off = 0;
  auto alloc = [&](size_t bytes) -> char* {
    char* p = ws + off;
    off += (bytes + 255) & ~(size_t)255;
    return p;
  };

  const size_t tokD = (size_t)NT_ * D_;      // 8M elements
  unsigned short* xb   = (unsigned short*)alloc(tokD * 2);
  unsigned short* encb = (unsigned short*)alloc(tokD * 2);
  unsigned short* Qb   = (unsigned short*)alloc(tokD * 2);
  unsigned short* Kb   = (unsigned short*)alloc(tokD * 2);
  unsigned short* Vb   = (unsigned short*)alloc(tokD * 2);
  unsigned short* zb   = (unsigned short*)alloc(tokD * 2);
  unsigned short* hb   = Qb;  // FFN hidden (8192x4096 bf16 = 64MB) reuses Q/K/V/z
  unsigned short* wq1t = (unsigned short*)alloc((size_t)D_ * H_ * DK_ * 2);
  unsigned short* wk1t = (unsigned short*)alloc((size_t)D_ * H_ * DK_ * 2);
  unsigned short* wv1t = (unsigned short*)alloc((size_t)D_ * H_ * DK_ * 2);
  unsigned short* wq2t = (unsigned short*)alloc((size_t)D_ * H_ * DK_ * 2);
  unsigned short* wk2t = (unsigned short*)alloc((size_t)D_ * H_ * DK_ * 2);
  unsigned short* wv2t = (unsigned short*)alloc((size_t)D_ * H_ * DK_ * 2);
  unsigned short* wo1t = (unsigned short*)alloc((size_t)D_ * D_ * 2);
  unsigned short* wo2t = (unsigned short*)alloc((size_t)D_ * D_ * 2);
  unsigned short* w1t  = (unsigned short*)alloc((size_t)D_ * DFF_ * 2);
  unsigned short* w2t  = (unsigned short*)alloc((size_t)DFF_ * D_ * 2);
  float* gemmf = (float*)alloc(tokD * 4);
  float* xf    = (float*)alloc(tokD * 4);

  auto blocks = [](size_t n) { return (unsigned)((n + 255) / 256); };

  // ---- pack inputs and (transposed) weights to bf16
  cvt_kernel<<<blocks(tokD), 256, 0, stream>>>(x, xb, tokD);
  cvt_kernel<<<blocks(tokD), 256, 0, stream>>>(enc, encb, tokD);
  const size_t qkvN = (size_t)H_ * D_ * DK_;
  pack_qkv_kernel<<<blocks(qkvN), 256, 0, stream>>>(Wq1, wq1t);
  pack_qkv_kernel<<<blocks(qkvN), 256, 0, stream>>>(Wk1, wk1t);
  pack_qkv_kernel<<<blocks(qkvN), 256, 0, stream>>>(Wv1, wv1t);
  pack_qkv_kernel<<<blocks(qkvN), 256, 0, stream>>>(Wq2, wq2t);
  pack_qkv_kernel<<<blocks(qkvN), 256, 0, stream>>>(Wk2, wk2t);
  pack_qkv_kernel<<<blocks(qkvN), 256, 0, stream>>>(Wv2, wv2t);
  cvt_t_kernel<<<blocks((size_t)D_ * D_), 256, 0, stream>>>(Wo1, wo1t, H_ * DK_, D_);
  cvt_t_kernel<<<blocks((size_t)D_ * D_), 256, 0, stream>>>(Wo2, wo2t, H_ * DK_, D_);
  cvt_t_kernel<<<blocks((size_t)D_ * DFF_), 256, 0, stream>>>(W1, w1t, D_, DFF_);
  cvt_t_kernel<<<blocks((size_t)DFF_ * D_), 256, 0, stream>>>(W2, w2t, DFF_, D_);

  const dim3 gProj(D_ / 128, NT_ / 128);         // (8, 64)
  const dim3 gFfn1(DFF_ / 128, NT_ / 128);       // (32, 64)
  const dim3 gAttn(L_ / 64, H_, B_);             // (16, 16, 8)

  // ---- self-attention (masked)
  gemm_kernel<0,0,1,0><<<gProj, 256, 0, stream>>>(xb, wq1t, bq1, nullptr, Qb, NT_, D_, D_);
  gemm_kernel<0,0,1,0><<<gProj, 256, 0, stream>>>(xb, wk1t, bk1, nullptr, Kb, NT_, D_, D_);
  gemm_kernel<0,0,0,1><<<gProj, 256, 0, stream>>>(xb, wv1t, bv1, nullptr, Vb, NT_, D_, D_);
  attn_kernel<true><<<gAttn, 128, 0, stream>>>(Qb, Kb, Vb, zb, L_);
  gemm_kernel<0,1,0,0><<<gProj, 256, 0, stream>>>(zb, wo1t, bo1, gemmf, nullptr, NT_, D_, D_);
  ln_kernel<<<NT_, 256, 0, stream>>>(gemmf, x, g1, be1, xf, xb);

  // ---- cross-attention (unmasked)
  gemm_kernel<0,0,1,0><<<gProj, 256, 0, stream>>>(xb,   wq2t, bq2, nullptr, Qb, NT_, D_, D_);
  gemm_kernel<0,0,1,0><<<gProj, 256, 0, stream>>>(encb, wk2t, bk2, nullptr, Kb, NT_, D_, D_);
  gemm_kernel<0,0,0,1><<<gProj, 256, 0, stream>>>(encb, wv2t, bv2, nullptr, Vb, NT_, D_, D_);
  attn_kernel<false><<<gAttn, 128, 0, stream>>>(Qb, Kb, Vb, zb, M_);
  gemm_kernel<0,1,0,0><<<gProj, 256, 0, stream>>>(zb, wo2t, bo2, gemmf, nullptr, NT_, D_, D_);
  ln_kernel<<<NT_, 256, 0, stream>>>(gemmf, xf, g2, be2, xf, xb);

  // ---- FFN (conv1d k=1 pair == per-token MLP)
  gemm_kernel<1,0,1,0><<<gFfn1, 256, 0, stream>>>(xb, w1t, b1, nullptr, hb, NT_, DFF_, D_);
  gemm_kernel<0,1,0,0><<<gProj, 256, 0, stream>>>(hb, w2t, b2, gemmf, nullptr, NT_, D_, DFF_);
  ln_kernel<<<NT_, 256, 0, stream>>>(gemmf, xf, g3, be3, (float*)d_out, nullptr);
}
